// RWKVAttention_54846732370009
// MI455X (gfx1250) — compile-verified, matrix operand order checked by reading, not run on hardware
//
#include <hip/hip_runtime.h>

#define BB 2
#define TT 2048
#define CC 2048
#define HH 16
#define DD 128
#define MM (BB*TT)          // 4096 rows
#define EPSLN 1e-5f

typedef __attribute__((ext_vector_type(16))) __bf16 v16bf;
typedef __attribute__((ext_vector_type(8)))  float  v8f;
typedef __attribute__((ext_vector_type(2)))  float  v2f;

union BfFrag { v16bf v; unsigned u[8]; };

static __device__ __forceinline__ unsigned short f2bf(float f) {
    union { float f; unsigned u; } cv; cv.f = f;
    unsigned r = (cv.u + 0x7FFFu + ((cv.u >> 16) & 1u)) >> 16;
    return (unsigned short)r;
}

// CDNA5 async global->LDS copy (16B per lane), ASYNCcnt-tracked.
// LDS dest passed as real pointer: low 32 bits of the generic address are the
// LDS byte offset (ISA 10.2), and the ptrtoint capture forces LLVM to treat
// the LDS object as written by the asm (keeps the ds_load fragment reads).
static __device__ __forceinline__ void async_cp16(unsigned short* ldsPtr,
                                                  const unsigned short* g) {
    unsigned lb = (unsigned)(uintptr_t)ldsPtr;
    unsigned long long ga = (unsigned long long)(uintptr_t)g;
    asm volatile("global_load_async_to_lds_b128 %0, %1, off"
                 :: "v"(lb), "v"(ga) : "memory");
}
static __device__ __forceinline__ void async_wait0() {
    asm volatile("s_wait_asynccnt 0x0" ::: "memory");
}

// ---------------------------------------------------------------------------
// Elementwise: token shift + 4 mixes, f32 -> bf16
// ---------------------------------------------------------------------------
__global__ __launch_bounds__(256) void mix_kernel(
    const float* __restrict__ x,
    const float* __restrict__ tmk, const float* __restrict__ tmv,
    const float* __restrict__ tmr, const float* __restrict__ tmg,
    unsigned short* __restrict__ xk, unsigned short* __restrict__ xv,
    unsigned short* __restrict__ xr, unsigned short* __restrict__ xg)
{
    int idx = blockIdx.x * 256 + threadIdx.x;
    if (idx >= BB*TT*CC) return;
    int c = idx % CC;
    int t = (idx / CC) % TT;
    int b = idx / (TT*CC);
    int tp = (t == 0) ? 1 : (t - 1);
    float xc = x[idx];
    float xx = x[((size_t)b*TT + tp)*CC + c] - xc;
    xk[idx] = f2bf(xc + xx * tmk[c]);
    xv[idx] = f2bf(xc + xx * tmv[c]);
    xr[idx] = f2bf(xc + xx * tmr[c]);
    xg[idx] = f2bf(xc + xx * tmg[c]);
}

__global__ __launch_bounds__(256) void wcvt_kernel(
    const float* __restrict__ w, unsigned short* __restrict__ o, int n)
{
    int idx = blockIdx.x * 256 + threadIdx.x;
    if (idx < n) o[idx] = f2bf(w[idx]);
}

// ---------------------------------------------------------------------------
// bf16 GEMM: out[M,N] = A[M,K] * W[N,K]^T, optional sigmoid epilogue.
// Block = 256 thr (8 waves). Block tile 128(M) x 64(N); wave tile 16x64.
// Double-buffered LDS staging via global_load_async_to_lds_b128 (ASYNCcnt),
// all fragments preloaded (clause of ds_load_b128, one dscnt wait), then
// 4 back-to-back v_wmma_f32_16x16x32_bf16 on independent accumulators.
// LDS per buffer: A 128x32 bf16 (8KB) + B 64x32 bf16 (4KB).
// ---------------------------------------------------------------------------
__global__ __launch_bounds__(256) void gemm_bf16_kernel(
    const unsigned short* __restrict__ A, const unsigned short* __restrict__ W,
    float* __restrict__ out, int M, int N, int K, int act)
{
    __shared__ unsigned short smem[2 * 6144];   // 2 x (4096 + 2048) halves = 24KB

    int tid  = threadIdx.x;
    int wave = tid >> 5;
    int lane = tid & 31;
    int m  = lane & 15;
    int g  = lane >> 4;
    int mrow0 = blockIdx.y * 128;
    int nblk  = blockIdx.x * 64;

    // ---- async tile copy issue (per thread: A 2x16B, B 1x16B) ----
    int ac_row0 =  tid >> 2;           // A rows 0..63 (and +64)
    int ac_part = (tid & 3) * 8;       // 8 halves = 16 bytes
    int bc_row  =  tid >> 2;           // B rows 0..63
    #define ISSUE_TILES(buf, kk)                                                   \
    {                                                                              \
        unsigned short* bufp = smem + (buf) * 6144;                                \
        async_cp16(bufp + ac_row0*32 + ac_part,                                    \
                   A + (size_t)(mrow0 + ac_row0) * K + (kk) + ac_part);            \
        async_cp16(bufp + (ac_row0 + 64)*32 + ac_part,                             \
                   A + (size_t)(mrow0 + ac_row0 + 64) * K + (kk) + ac_part);       \
        async_cp16(bufp + 4096 + bc_row*32 + ac_part,                              \
                   W + (size_t)(nblk + bc_row) * K + (kk) + ac_part);              \
    }

    v8f acc[4] = {};
    ISSUE_TILES(0, 0);
    async_wait0();
    __syncthreads();

    int cur = 0;
    int arow = wave*16 + m;
    for (int kk = 0; kk < K; kk += 32) {
        int nxt = cur ^ 1;
        if (kk + 32 < K) ISSUE_TILES(nxt, kk + 32);

        const unsigned short* as = smem + cur * 6144;
        const unsigned short* bs = as + 4096;

        // Preload A + all 4 B fragments, then run the WMMA chain.
        BfFrag au, bu[4];
        *(uint4*)&au.u[0] = *(const uint4*)(as + arow*32 + g*8);
        *(uint4*)&au.u[4] = *(const uint4*)(as + arow*32 + 16 + g*8);
        #pragma unroll
        for (int s = 0; s < 4; ++s) {
            int brow = s*16 + m;
            *(uint4*)&bu[s].u[0] = *(const uint4*)(bs + brow*32 + g*8);
            *(uint4*)&bu[s].u[4] = *(const uint4*)(bs + brow*32 + 16 + g*8);
        }
        #pragma unroll
        for (int s = 0; s < 4; ++s)
            acc[s] = __builtin_amdgcn_wmma_f32_16x16x32_bf16(
                false, au.v, false, bu[s].v, (short)0, acc[s], false, false);

        async_wait0();       // next tiles landed
        __syncthreads();     // everyone done reading cur
        cur = nxt;
    }
    #undef ISSUE_TILES

    // C/D layout: lane col n=l&15, VGPR i -> row i + 8*(l>>4)
    int mblk = mrow0 + wave * 16;
    #pragma unroll
    for (int s = 0; s < 4; ++s)
        #pragma unroll
        for (int i = 0; i < 8; ++i) {
            int row = mblk + i + 8*g;
            int col = nblk + s*16 + m;
            float v = acc[s][i];
            if (act) v = 1.0f / (1.0f + __expf(-v));
            out[(size_t)row * N + col] = v;
        }
}

// ---------------------------------------------------------------------------
// Chunked WKV scan, chunk L=16, one workgroup per (b,h), f32 WMMA 16x16x4.
//   R~[t,i] = r[t,i]*w_i^t ; K~[s,i] = k[s,i]*w_i^-(s+1)
//   scores A[t,s] = R~ . K~^T  (mask s<t, diag = sum_i r u k)
//   O = A_mask.V + R~.S0 ;  S <- w^16 * (S0 + K~^T . V)
// ---------------------------------------------------------------------------
__global__ __launch_bounds__(256) void wkv_kernel(
    const float* __restrict__ r, const float* __restrict__ k,
    const float* __restrict__ v,
    const float* __restrict__ td, const float* __restrict__ tf,
    float* __restrict__ o)
{
    extern __shared__ float lds[];
    float* S   = lds;               // 128*128
    float* rt  = S   + 16384;       // 16*128  (R~)
    float* kt  = rt  + 2048;        // 16*128  (K~)
    float* vt  = kt  + 2048;        // 16*128  (V)
    float* sc  = vt  + 2048;        // 16*16   masked scores
    float* wp  = sc  + 256;         // 17*128  w^t
    float* wpi = wp  + 2176;        // 17*128  w^-t
    float* uu  = wpi + 2176;        // 128
    float* dgp = uu  + 128;         // 128 diag partials

    int b = blockIdx.x / HH;
    int h = blockIdx.x % HH;
    int tid  = threadIdx.x;
    int wave = tid >> 5;
    int lane = tid & 31;
    int lm = lane & 15;
    int lg = lane >> 4;

    if (tid < 128) {
        float w  = __expf(-__expf(td[h*DD + tid]));
        float wi = 1.0f / w;
        uu[tid] = tf[h*DD + tid];
        float p = 1.0f, q = 1.0f;
        for (int t = 0; t <= 16; ++t) {
            wp[t*128 + tid] = p; wpi[t*128 + tid] = q;
            p *= w; q *= wi;
        }
    }
    for (int i = tid; i < 16384; i += 256) S[i] = 0.0f;
    __syncthreads();

    const size_t base = (size_t)b * TT * CC + h * DD;

    for (int ch = 0; ch < TT/16; ++ch) {
        // --- stage + scale chunk into LDS ---
        for (int e = tid; e < 2048; e += 256) {
            int t = e >> 7, i = e & 127;
            size_t gidx = base + (size_t)(ch*16 + t) * CC + i;
            rt[e] = r[gidx] * wp[t*128 + i];
            kt[e] = k[gidx] * wpi[(t+1)*128 + i];
            vt[e] = v[gidx];
        }
        if (tid < 128) {        // diag: sum_i r[t,i]*u[i]*k[t,i], 8 partials/t
            int t = tid >> 3, p8 = tid & 7;
            float s = 0.0f;
            size_t gt = base + (size_t)(ch*16 + t) * CC;
            for (int i = p8*16; i < p8*16 + 16; ++i)
                s += r[gt + i] * uu[i] * k[gt + i];
            dgp[tid] = s;
        }
        __syncthreads();

        // --- scores (wave 0): A = R~ . K~^T, 16x16, K=128 ---
        if (wave == 0) {
            v8f a = {};
            for (int kb = 0; kb < 128; kb += 4) {
                v2f af, bf;
                af.x = rt[lm*128 + kb + lg*2];
                af.y = rt[lm*128 + kb + lg*2 + 1];
                bf.x = kt[lm*128 + kb + lg*2];        // B[i][s] = K~[s][i], s=lm
                bf.y = kt[lm*128 + kb + lg*2 + 1];
                a = __builtin_amdgcn_wmma_f32_16x16x4_f32(
                    false, af, false, bf, (short)0, a, false, false);
            }
            #pragma unroll
            for (int i = 0; i < 8; ++i) {
                int t = i + 8*lg, scol = lm;
                float val = a[i];
                if (scol > t) val = 0.0f;
                else if (scol == t) {
                    val = 0.0f;
                    for (int pp = 0; pp < 8; ++pp) val += dgp[t*8 + pp];
                }
                sc[t*16 + scol] = val;
            }
        }
        __syncthreads();

        // --- per wave: output N-tile + state row-slab ---
        int nb = wave * 16;
        v8f oacc = {};
        for (int kb = 0; kb < 128; kb += 4) {       // inter: R~ . S0
            v2f af, bf;
            af.x = rt[lm*128 + kb + lg*2];
            af.y = rt[lm*128 + kb + lg*2 + 1];
            bf.x = S[(kb + lg*2    )*128 + nb + lm];
            bf.y = S[(kb + lg*2 + 1)*128 + nb + lm];
            oacc = __builtin_amdgcn_wmma_f32_16x16x4_f32(
                false, af, false, bf, (short)0, oacc, false, false);
        }
        for (int kb = 0; kb < 16; kb += 4) {        // intra: A_mask . V
            v2f af, bf;
            af.x = sc[lm*16 + kb + lg*2];
            af.y = sc[lm*16 + kb + lg*2 + 1];
            bf.x = vt[(kb + lg*2    )*128 + nb + lm];
            bf.y = vt[(kb + lg*2 + 1)*128 + nb + lm];
            oacc = __builtin_amdgcn_wmma_f32_16x16x4_f32(
                false, af, false, bf, (short)0, oacc, false, false);
        }
        // state: wave handles rows [wave*16, +16) of S, all 8 col-tiles
        int mb = wave * 16;
        v8f sacc[8];
        #pragma unroll
        for (int nt = 0; nt < 8; ++nt) {
            int nb2 = nt * 16;
            v8f cc;
            #pragma unroll
            for (int i = 0; i < 8; ++i)
                cc[i] = S[(mb + i + 8*lg)*128 + nb2 + lm];
            for (int kb = 0; kb < 16; kb += 4) {    // K~^T . V  (A[m][s]=K~[s][mb+m])
                v2f af, bf;
                af.x = kt[(kb + lg*2    )*128 + mb + lm];
                af.y = kt[(kb + lg*2 + 1)*128 + mb + lm];
                bf.x = vt[(kb + lg*2    )*128 + nb2 + lm];
                bf.y = vt[(kb + lg*2 + 1)*128 + nb2 + lm];
                cc = __builtin_amdgcn_wmma_f32_16x16x4_f32(
                    false, af, false, bf, (short)0, cc, false, false);
            }
            sacc[nt] = cc;
        }
        __syncthreads();                 // all reads of S done
        #pragma unroll
        for (int nt = 0; nt < 8; ++nt)
            #pragma unroll
            for (int i = 0; i < 8; ++i) {
                int rrow = mb + i + 8*lg;
                S[rrow*128 + nt*16 + lm] = sacc[nt][i] * wp[16*128 + rrow];
            }
        #pragma unroll
        for (int i = 0; i < 8; ++i) {
            int t = i + 8*lg;
            o[base + (size_t)(ch*16 + t)*CC + nb + lm] = oacc[i];
        }
        __syncthreads();
    }
}

// ---------------------------------------------------------------------------
// og = o * sigmoid_g (sigmoid applied in g-GEMM epilogue) -> bf16
// ---------------------------------------------------------------------------
__global__ __launch_bounds__(256) void og_kernel(
    const float* __restrict__ o, const float* __restrict__ g,
    unsigned short* __restrict__ og, int n)
{
    int idx = blockIdx.x * 256 + threadIdx.x;
    if (idx < n) og[idx] = f2bf(o[idx] * g[idx]);
}

// ---------------------------------------------------------------------------
// In-place LayerNorm over last dim (2048), one block per row
// ---------------------------------------------------------------------------
__global__ __launch_bounds__(256) void ln_kernel(
    float* __restrict__ y, const float* __restrict__ w, const float* __restrict__ bb)
{
    __shared__ float red[256];
    float* yr = y + (size_t)blockIdx.x * CC;
    float vals[8];
    float s = 0.0f;
    #pragma unroll
    for (int i = 0; i < 8; ++i) { vals[i] = yr[threadIdx.x + i*256]; s += vals[i]; }
    red[threadIdx.x] = s; __syncthreads();
    for (int st = 128; st > 0; st >>= 1) {
        if (threadIdx.x < st) red[threadIdx.x] += red[threadIdx.x + st];
        __syncthreads();
    }
    float mu = red[0] * (1.0f / CC);
    __syncthreads();
    float s2 = 0.0f;
    #pragma unroll
    for (int i = 0; i < 8; ++i) { float d = vals[i] - mu; s2 += d*d; }
    red[threadIdx.x] = s2; __syncthreads();
    for (int st = 128; st > 0; st >>= 1) {
        if (threadIdx.x < st) red[threadIdx.x] += red[threadIdx.x + st];
        __syncthreads();
    }
    float inv = rsqrtf(red[0] * (1.0f / CC) + EPSLN);
    #pragma unroll
    for (int i = 0; i < 8; ++i) {
        int c = threadIdx.x + i*256;
        yr[c] = (vals[i] - mu) * inv * w[c] + bb[c];
    }
}

// ---------------------------------------------------------------------------
extern "C" void kernel_launch(void* const* d_in, const int* in_sizes, int n_in,
                              void* d_out, int out_size, void* d_ws, size_t ws_size,
                              hipStream_t stream)
{
    const float* x   = (const float*)d_in[0];
    const float* Wr  = (const float*)d_in[1];
    const float* Wk  = (const float*)d_in[2];
    const float* Wv  = (const float*)d_in[3];
    const float* Wg  = (const float*)d_in[4];
    const float* Wo  = (const float*)d_in[5];
    const float* td  = (const float*)d_in[6];
    const float* tf  = (const float*)d_in[7];
    const float* tmk = (const float*)d_in[8];
    const float* tmv = (const float*)d_in[9];
    const float* tmr = (const float*)d_in[10];
    const float* tmg = (const float*)d_in[11];
    const float* lnw = (const float*)d_in[12];
    const float* lnb = (const float*)d_in[13];
    float* yout = (float*)d_out;

    char* ws = (char*)d_ws;
    const size_t MB = 1u << 20;
    unsigned short* wrb = (unsigned short*)(ws + 0*MB);      // 8 MB each
    unsigned short* wkb = (unsigned short*)(ws + 8*MB);
    unsigned short* wvb = (unsigned short*)(ws + 16*MB);
    unsigned short* wgb = (unsigned short*)(ws + 24*MB);
    unsigned short* wob = (unsigned short*)(ws + 32*MB);
    unsigned short* xrb = (unsigned short*)(ws + 40*MB);     // 16 MB each
    unsigned short* xkb = (unsigned short*)(ws + 56*MB);
    unsigned short* xvb = (unsigned short*)(ws + 72*MB);
    unsigned short* xgb = (unsigned short*)(ws + 88*MB);
    float* rbuf = (float*)(ws + 104*MB);                     // 32 MB each
    float* kbuf = (float*)(ws + 136*MB);
    float* vbuf = (float*)(ws + 168*MB);
    float* gbuf = (float*)(ws + 200*MB);
    float* obuf = (float*)(ws + 232*MB);
    unsigned short* ogb = (unsigned short*)(ws + 264*MB);    // 16 MB

    const int NEL = BB*TT*CC;           // 8M
    const int WEL = CC*CC;              // 4M

    wcvt_kernel<<<(WEL+255)/256, 256, 0, stream>>>(Wr, wrb, WEL);
    wcvt_kernel<<<(WEL+255)/256, 256, 0, stream>>>(Wk, wkb, WEL);
    wcvt_kernel<<<(WEL+255)/256, 256, 0, stream>>>(Wv, wvb, WEL);
    wcvt_kernel<<<(WEL+255)/256, 256, 0, stream>>>(Wg, wgb, WEL);
    wcvt_kernel<<<(WEL+255)/256, 256, 0, stream>>>(Wo, wob, WEL);

    mix_kernel<<<(NEL+255)/256, 256, 0, stream>>>(x, tmk, tmv, tmr, tmg,
                                                  xkb, xvb, xrb, xgb);

    dim3 gg(CC/64, MM/128);
    gemm_bf16_kernel<<<gg, 256, 0, stream>>>(xrb, wrb, rbuf, MM, CC, CC, 0);
    gemm_bf16_kernel<<<gg, 256, 0, stream>>>(xkb, wkb, kbuf, MM, CC, CC, 0);
    gemm_bf16_kernel<<<gg, 256, 0, stream>>>(xvb, wvb, vbuf, MM, CC, CC, 0);
    gemm_bf16_kernel<<<gg, 256, 0, stream>>>(xgb, wgb, gbuf, MM, CC, CC, 1);

    size_t shbytes = (16384 + 3*2048 + 256 + 2*2176 + 128 + 128) * sizeof(float);
    wkv_kernel<<<BB*HH, 256, shbytes, stream>>>(rbuf, kbuf, vbuf, td, tf, obuf);

    og_kernel<<<(NEL+255)/256, 256, 0, stream>>>(obuf, gbuf, ogb, NEL);

    gemm_bf16_kernel<<<gg, 256, 0, stream>>>(ogb, wob, yout, MM, CC, CC, 0);

    ln_kernel<<<MM, 256, 0, stream>>>(yout, lnw, lnb);
}
